// MossFormer_34677565948396
// MI455X (gfx1250) — matrix operand order repaired and checked by reading, not compiled, not run on hardware
//
#include <hip/hip_runtime.h>
#include <math.h>

// ---------------------------------------------------------------------------
// MossFormer block for MI455X (gfx1250). All heavy GEMMs run on
// v_wmma_f32_16x16x32_bf16 (bf16 in, f32 accum) with a 2-deep register
// pipeline (load K-step k+32 while WMMAs consume K-step k). Everything else
// (LN, shift, depthwise conv K=17, rotary, cross-chunk T=64 attention,
// gating) is VALU.
//
// Shapes: B=2 T=64 Q=256 C=512 -> BT=128 rows-of-256 tokens; H=2048, QK=128.
// ---------------------------------------------------------------------------

typedef __attribute__((ext_vector_type(16))) __bf16        bf16x16;
typedef __attribute__((ext_vector_type(8)))  float         f32x8;
typedef __attribute__((ext_vector_type(4)))  unsigned int  u32x4;

union Frag {
  u32x4   q[2];
  bf16x16 v;
};

__device__ __forceinline__ unsigned short f2bf(float f) {
  union { float f; unsigned int u; } cv;
  cv.f = f;
  unsigned int u = cv.u;
  u += 0x7FFFu + ((u >> 16) & 1u);   // round-to-nearest-even
  return (unsigned short)(u >> 16);
}

// ---------------------------------------------------------------------------
// Batched WMMA GEMM: C[b] = act(alpha * A[b] @ B[b] + bias)  (+= when ACC=1)
// A: bf16 row-major MxK.  B: bf16 pre-packed: Bp[((k/32)*N + n)*32 + k%32].
// Block: 256 threads = 8 waves (4 M-waves x 2 N-waves); block tile 64x128,
// wave tile 16x64 (4 wmma accumulators). Requires M%64==0, N%128==0, K%64==0.
// Double-buffered fragment registers: loads for K-step k+32 are issued before
// the WMMAs of K-step k, so waits become partial instead of loadcnt-0 drains.
// ACT: 0=none, 1=SiLU, 2=ReLU^2.
// ---------------------------------------------------------------------------
template <int ACT, int ACC>
__global__ void __launch_bounds__(256)
k_gemm(const unsigned short* __restrict__ A, long long sA,
       const unsigned short* __restrict__ B, long long sB,
       const float* __restrict__ bias,
       float* __restrict__ C, long long sC,
       int M, int Kd, int N, float alpha) {
  const int b = blockIdx.z;
  const unsigned short* Ab = A + sA * (long long)b;
  const unsigned short* Bb = B + sB * (long long)b;
  float*                Cb = C + sC * (long long)b;

  const int lane = threadIdx.x & 31;
  const int wave = threadIdx.x >> 5;
  const int m0 = blockIdx.y * 64 + (wave & 3) * 16;
  const int n0 = blockIdx.x * 128 + (wave >> 2) * 64;

  // A fragment addressing (16-bit A 16x32 layout, ISA 7.12.2)
  const int mrow = m0 + (lane & 15);
  const int kbA  = (lane < 16) ? 0 : 8;
  // B fragment addressing (16-bit B 32x16 layout)
  const int nlo  = lane & 15;
  const int kbB  = (lane < 16) ? 0 : 8;

  f32x8 acc[4];
#pragma unroll
  for (int t = 0; t < 4; ++t)
#pragma unroll
    for (int r = 0; r < 8; ++r) acc[t][r] = 0.0f;

  const unsigned short* arow = Ab + (long long)mrow * Kd;

  Frag fa0, fa1;
  Frag fb0[4], fb1[4];

  auto ldA = [&](Frag& f, int k0) {
    f.q[0] = *reinterpret_cast<const u32x4*>(arow + k0 + kbA);        // K kb..kb+7
    f.q[1] = *reinterpret_cast<const u32x4*>(arow + k0 + kbA + 16);   // K kb+16..kb+23
  };
  auto ldB = [&](Frag* f, int k0) {
    const unsigned short* bblk = Bb + ((long long)(k0 >> 5) * N) * 32;
#pragma unroll
    for (int t = 0; t < 4; ++t) {
      const unsigned short* bp = bblk + (long long)(n0 + t * 16 + nlo) * 32 + kbB;
      f[t].q[0] = *reinterpret_cast<const u32x4*>(bp);       // K kb2..kb2+7
      f[t].q[1] = *reinterpret_cast<const u32x4*>(bp + 16);  // K kb2+16..kb2+23
    }
  };

  // prologue: fill buffer 0
  ldA(fa0, 0);
  ldB(fb0, 0);

  for (int k0 = 0; k0 < Kd; k0 += 64) {
    // prefetch buffer 1 (k0+32 always valid: Kd % 64 == 0)
    ldA(fa1, k0 + 32);
    ldB(fb1, k0 + 32);
#pragma unroll
    for (int t = 0; t < 4; ++t)
      acc[t] = __builtin_amdgcn_wmma_f32_16x16x32_bf16(
          false, fa0.v, false, fb0[t].v, (short)0, acc[t], false, false);
    // prefetch buffer 0 for next iteration
    if (k0 + 64 < Kd) {
      ldA(fa0, k0 + 64);
      ldB(fb0, k0 + 64);
    }
#pragma unroll
    for (int t = 0; t < 4; ++t)
      acc[t] = __builtin_amdgcn_wmma_f32_16x16x32_bf16(
          false, fa1.v, false, fb1[t].v, (short)0, acc[t], false, false);
  }

#pragma unroll
  for (int t = 0; t < 4; ++t) {
    const int n  = n0 + t * 16 + nlo;
    const float bv = bias ? bias[n] : 0.0f;
#pragma unroll
    for (int r = 0; r < 8; ++r) {
      const int row = m0 + ((lane < 16) ? r : r + 8);
      float v = alpha * acc[t][r] + bv;
      if (ACT == 1) v = v / (1.0f + __expf(-v));             // SiLU
      if (ACT == 2) { v = v > 0.0f ? v : 0.0f; v = v * v; }  // ReLU^2
      const long long off = (long long)row * N + n;
      if (ACC) Cb[off] += v; else Cb[off] = v;
    }
  }
}

// ---------------------------------------------------------------------------
// Token-shift + concat + LayerNorm(C=512) -> two bf16 activation tensors
// (th and qk affine params differ). One block per row (32768 rows).
// ---------------------------------------------------------------------------
__global__ void __launch_bounds__(256)
k_ln_shift(const float* __restrict__ x,
           const float* __restrict__ tg, const float* __restrict__ tb,
           const float* __restrict__ qg, const float* __restrict__ qb,
           unsigned short* __restrict__ Ath, unsigned short* __restrict__ Aqk) {
  const long long r = blockIdx.x;
  const int c = threadIdx.x;         // 0..255, handles c and c+256
  const int qpos = (int)(r & 255);
  const float v0 = (qpos == 0) ? 0.0f : x[(r - 1) * 512 + c];   // shifted half
  const float v1 = x[r * 512 + 256 + c];                        // pass half

  __shared__ float sbuf[256];
  sbuf[c] = v0 + v1;
  __syncthreads();
  for (int o = 128; o > 0; o >>= 1) { if (c < o) sbuf[c] += sbuf[c + o]; __syncthreads(); }
  const float mean = sbuf[0] * (1.0f / 512.0f);
  __syncthreads();
  const float d0 = v0 - mean, d1 = v1 - mean;
  sbuf[c] = d0 * d0 + d1 * d1;
  __syncthreads();
  for (int o = 128; o > 0; o >>= 1) { if (c < o) sbuf[c] += sbuf[c + o]; __syncthreads(); }
  const float rstd = rsqrtf(sbuf[0] * (1.0f / 512.0f) + 1e-5f);

  const float n0 = d0 * rstd, n1 = d1 * rstd;
  Ath[r * 512 + c]       = f2bf(n0 * tg[c] + tb[c]);
  Ath[r * 512 + 256 + c] = f2bf(n1 * tg[256 + c] + tb[256 + c]);
  Aqk[r * 512 + c]       = f2bf(n0 * qg[c] + qb[c]);
  Aqk[r * 512 + 256 + c] = f2bf(n1 * qg[256 + c] + qb[256 + c]);
}

// ---------------------------------------------------------------------------
// Depthwise conv along Q (K=17, pad 8) + identity (+ optional residual):
// dst = (res?res:0) + src + conv(src). w layout: (C,1,17).
// ---------------------------------------------------------------------------
__global__ void __launch_bounds__(256)
k_dwconv(const float* __restrict__ src, const float* __restrict__ w,
         const float* __restrict__ res, float* __restrict__ dst,
         int C, long long total) {
  const long long idx = (long long)blockIdx.x * 256 + threadIdx.x;
  if (idx >= total) return;
  const int c = (int)(idx % C);
  const long long r = idx / C;
  const int qpos = (int)(r & 255);
  const long long base = r - qpos;   // start of this sequence
  float acc = src[idx];
#pragma unroll
  for (int k = 0; k < 17; ++k) {
    const int qq = qpos + k - 8;
    if (qq >= 0 && qq < 256) acc += src[(base + qq) * C + c] * w[c * 17 + k];
  }
  if (res) acc += res[idx];
  dst[idx] = acc;
}

// ---------------------------------------------------------------------------
// qs = qkhid*gamma[i]+beta[i], rotary on first 32 dims, route 4 heads:
// i=0 quad_q (f32 + bf16 A), i=1 lin_q (bf16 A), i=2 quad_k (f32),
// i=3 lin_k stored TRANSPOSED per batch (bf16 A for lin_k^T @ hid).
// One block (128 threads) per row.
// ---------------------------------------------------------------------------
__global__ void __launch_bounds__(128)
k_rotary(const float* __restrict__ qkhid,
         const float* __restrict__ gamma, const float* __restrict__ beta,
         float* __restrict__ qqf, unsigned short* __restrict__ qqbf,
         unsigned short* __restrict__ lqbf,
         float* __restrict__ qkf, unsigned short* __restrict__ lkT) {
  const long long r = blockIdx.x;
  const int d = threadIdx.x;           // 0..127
  const int qpos = (int)(r & 255);
  const int bt = (int)(r >> 8);

  const float s0 = qkhid[r * 128 + d];
  float cosv = 1.0f, sinv = 0.0f, sp0 = 0.0f;
  const bool rot = d < 32;
  if (rot) {
    const int p = d >> 1;
    const float invf = powf(10000.0f, -(float)(2 * p) / 32.0f);
    const float ang = (float)qpos * invf;
    cosv = cosf(ang);
    sinv = sinf(ang);
    sp0 = qkhid[r * 128 + (d ^ 1)];
  }
#pragma unroll
  for (int i = 0; i < 4; ++i) {
    float v = s0 * gamma[i * 128 + d] + beta[i * 128 + d];
    if (rot) {
      const float sp = sp0 * gamma[i * 128 + (d ^ 1)] + beta[i * 128 + (d ^ 1)];
      v = (d & 1) ? (v * cosv + sp * sinv) : (v * cosv - sp * sinv);
    }
    if (i == 0)      { qqf[r * 128 + d] = v; qqbf[r * 128 + d] = f2bf(v); }
    else if (i == 1) { lqbf[r * 128 + d] = f2bf(v); }
    else if (i == 2) { qkf[r * 128 + d] = v; }
    else             { lkT[(long long)bt * 32768 + (long long)d * 256 + qpos] = f2bf(v); }
  }
}

// pack row-major f32 src (batched KxN) into bf16 WMMA B layout
__global__ void __launch_bounds__(256)
k_pack_B(const float* __restrict__ src, unsigned short* __restrict__ dst,
         int K, int N, long long total) {
  const long long t = (long long)blockIdx.x * 256 + threadIdx.x;
  if (t >= total) return;
  const long long per = (long long)K * N;
  const long long b = t / per;
  const long long rem = t - b * per;
  const int k = (int)(rem / N);
  const int n = (int)(rem % N);
  dst[b * per + ((long long)(k >> 5) * N + n) * 32 + (k & 31)] = f2bf(src[t]);
}

// pack f32 src given as NxK row-major (i.e. B = src^T, logical KxN)
__global__ void __launch_bounds__(256)
k_pack_BT(const float* __restrict__ src, unsigned short* __restrict__ dst,
          int K, int N, long long total) {
  const long long t = (long long)blockIdx.x * 256 + threadIdx.x;
  if (t >= total) return;
  const long long per = (long long)K * N;
  const long long b = t / per;
  const long long rem = t - b * per;
  const int n = (int)(rem / K);   // src row
  const int k = (int)(rem % K);   // src col
  dst[b * per + ((long long)(k >> 5) * N + n) * 32 + (k & 31)] = f2bf(src[t]);
}

__global__ void __launch_bounds__(256)
k_cvt_bf16(const float* __restrict__ src, unsigned short* __restrict__ dst,
           long long total) {
  const long long t = (long long)blockIdx.x * 256 + threadIdx.x;
  if (t < total) dst[t] = f2bf(src[t]);
}

// ---------------------------------------------------------------------------
// Cross-chunk attention over T=64 (diag masked), accumulates into attuv.
// Block = one (b, qh, i) output row. <5% of FLOPs -> VALU is fine.
// ---------------------------------------------------------------------------
__global__ void __launch_bounds__(256)
k_cross(const float* __restrict__ qqf, const float* __restrict__ qkf,
        const float* __restrict__ hid, float* __restrict__ attuv) {
  const int i  = blockIdx.x;   // 0..63
  const int qh = blockIdx.y;   // 0..255
  const int b  = blockIdx.z;   // 0..1
  const int tid = threadIdx.x;
  __shared__ float sa[64];

  const long long rowi = ((long long)(b * 64 + i) * 256 + qh);
  if (tid < 64) {
    const int j = tid;
    const long long rowj = ((long long)(b * 64 + j) * 256 + qh);
    const float* qp = qqf + rowi * 128;
    const float* kp = qkf + rowj * 128;
    float acc = 0.0f;
    for (int d = 0; d < 128; ++d) acc += qp[d] * kp[d];
    acc *= (1.0f / 64.0f);
    acc = acc > 0.0f ? acc : 0.0f;
    acc *= acc;
    if (j == i) acc = 0.0f;     // zero diagonal
    sa[j] = acc;
  }
  __syncthreads();
  for (int e = tid; e < 2048; e += 256) {
    float acc = 0.0f;
    for (int j = 0; j < 64; ++j)
      acc += sa[j] * hid[((long long)(b * 64 + j) * 256 + qh) * 2048 + e];
    attuv[rowi * 2048 + e] += acc;
  }
}

// ---------------------------------------------------------------------------
// Gating out = att_u * v * sigmoid(att_v * u), then LayerNorm(1024) -> bf16 A.
// attuv = [att_v | att_u], hid = [v | u]. One block per row.
// ---------------------------------------------------------------------------
__global__ void __launch_bounds__(256)
k_gate_ln(const float* __restrict__ attuv, const float* __restrict__ hid,
          const float* __restrict__ g, const float* __restrict__ bb,
          unsigned short* __restrict__ Aout) {
  const long long r = blockIdx.x;
  const int tid = threadIdx.x;
  __shared__ float sbuf[256];

  float o[4];
  float s = 0.0f;
#pragma unroll
  for (int t = 0; t < 4; ++t) {
    const int e = tid + t * 256;
    const float av = attuv[r * 2048 + e];
    const float au = attuv[r * 2048 + 1024 + e];
    const float v  = hid[r * 2048 + e];
    const float u  = hid[r * 2048 + 1024 + e];
    const float ov = au * v * (1.0f / (1.0f + __expf(-av * u)));
    o[t] = ov;
    s += ov;
  }
  sbuf[tid] = s;
  __syncthreads();
  for (int ofs = 128; ofs > 0; ofs >>= 1) { if (tid < ofs) sbuf[tid] += sbuf[tid + ofs]; __syncthreads(); }
  const float mean = sbuf[0] * (1.0f / 1024.0f);
  __syncthreads();
  s = 0.0f;
#pragma unroll
  for (int t = 0; t < 4; ++t) { const float dd = o[t] - mean; s += dd * dd; }
  sbuf[tid] = s;
  __syncthreads();
  for (int ofs = 128; ofs > 0; ofs >>= 1) { if (tid < ofs) sbuf[tid] += sbuf[tid + ofs]; __syncthreads(); }
  const float rstd = rsqrtf(sbuf[0] * (1.0f / 1024.0f) + 1e-5f);
  __syncthreads();
#pragma unroll
  for (int t = 0; t < 4; ++t) {
    const int e = tid + t * 256;
    Aout[r * 1024 + e] = f2bf((o[t] - mean) * rstd * g[e] + bb[e]);
  }
}

// ---------------------------------------------------------------------------
extern "C" void kernel_launch(void* const* d_in, const int* in_sizes, int n_in,
                              void* d_out, int out_size, void* d_ws, size_t ws_size,
                              hipStream_t stream) {
  (void)in_sizes; (void)n_in; (void)out_size; (void)ws_size;
  const float* x       = (const float*)d_in[0];
  const float* th_g    = (const float*)d_in[1];
  const float* th_bln  = (const float*)d_in[2];
  const float* th_W    = (const float*)d_in[3];
  const float* th_b    = (const float*)d_in[4];
  const float* th_conv = (const float*)d_in[5];
  const float* qk_g    = (const float*)d_in[6];
  const float* qk_bln  = (const float*)d_in[7];
  const float* qk_W    = (const float*)d_in[8];
  const float* qk_b    = (const float*)d_in[9];
  const float* qk_conv = (const float*)d_in[10];
  const float* gamma   = (const float*)d_in[11];
  const float* beta    = (const float*)d_in[12];
  const float* o_g     = (const float*)d_in[13];
  const float* o_bln   = (const float*)d_in[14];
  const float* out_W   = (const float*)d_in[15];
  const float* out_b   = (const float*)d_in[16];
  const float* o_conv  = (const float*)d_in[17];
  float* out = (float*)d_out;

  const long long ROWS = 32768;  // (B*T)*Q
  char* wp = (char*)d_ws;
  auto alloc = [&](size_t bytes) {
    char* p = wp;
    wp += (bytes + 255) & ~(size_t)255;
    return p;
  };
  unsigned short* nAth   = (unsigned short*)alloc(ROWS * 512 * 2);
  unsigned short* nAqk   = (unsigned short*)alloc(ROWS * 512 * 2);
  unsigned short* thWp   = (unsigned short*)alloc(512 * 2048 * 2);
  unsigned short* qkWp   = (unsigned short*)alloc(512 * 128 * 2);
  unsigned short* outWp  = (unsigned short*)alloc(1024 * 512 * 2);
  float*          h      = (float*)alloc(ROWS * 2048 * 4);   // reused as attuv
  float*          hid    = (float*)alloc(ROWS * 2048 * 4);
  unsigned short* hid_p  = (unsigned short*)alloc(ROWS * 2048 * 2);
  float*          qkh    = (float*)alloc(ROWS * 128 * 4);
  float*          qkhid  = (float*)alloc(ROWS * 128 * 4);
  float*          qqf    = (float*)alloc(ROWS * 128 * 4);
  float*          qkf    = (float*)alloc(ROWS * 128 * 4);
  unsigned short* qqbf   = (unsigned short*)alloc(ROWS * 128 * 2);
  unsigned short* lqbf   = (unsigned short*)alloc(ROWS * 128 * 2);
  unsigned short* lkT    = (unsigned short*)alloc(ROWS * 128 * 2);
  unsigned short* qk_p   = (unsigned short*)alloc(ROWS * 128 * 2);
  float*          attn   = (float*)alloc(128LL * 256 * 256 * 4);
  unsigned short* attnbf = (unsigned short*)alloc(128LL * 256 * 256 * 2);
  float*          linkv  = (float*)alloc(128LL * 128 * 2048 * 4);
  unsigned short* linkvp = (unsigned short*)alloc(128LL * 128 * 2048 * 2);
  unsigned short* Aout   = (unsigned short*)alloc(ROWS * 1024 * 2);
  float*          oh     = (float*)alloc(ROWS * 512 * 4);
  float* attuv = h;   // h is dead after hid is formed

  auto blocks = [](long long total) { return (unsigned)((total + 255) / 256); };

  // ---- weight packing (bf16 WMMA B layout) ----
  k_pack_B<<<blocks(512LL * 2048), 256, 0, stream>>>(th_W, thWp, 512, 2048, 512LL * 2048);
  k_pack_B<<<blocks(512LL * 128), 256, 0, stream>>>(qk_W, qkWp, 512, 128, 512LL * 128);
  k_pack_B<<<blocks(1024LL * 512), 256, 0, stream>>>(out_W, outWp, 1024, 512, 1024LL * 512);

  // ---- token-shift + LN -> bf16 activations ----
  k_ln_shift<<<(unsigned)ROWS, 256, 0, stream>>>(x, th_g, th_bln, qk_g, qk_bln, nAth, nAqk);

  // ---- th FFN: h = SiLU(LN @ W + b); hid = h + dwconv(h) ----
  k_gemm<1, 0><<<dim3(2048 / 128, (unsigned)(ROWS / 64), 1), 256, 0, stream>>>(
      nAth, 0, thWp, 0, th_b, h, 0, (int)ROWS, 512, 2048, 1.0f);
  k_dwconv<<<blocks(ROWS * 2048), 256, 0, stream>>>(h, th_conv, nullptr, hid, 2048, ROWS * 2048);
  k_pack_B<<<blocks(ROWS * 2048), 256, 0, stream>>>(hid, hid_p, 256, 2048, ROWS * 2048);

  // ---- qk path ----
  k_gemm<1, 0><<<dim3(128 / 128, (unsigned)(ROWS / 64), 1), 256, 0, stream>>>(
      nAqk, 0, qkWp, 0, qk_b, qkh, 0, (int)ROWS, 512, 128, 1.0f);
  k_dwconv<<<blocks(ROWS * 128), 256, 0, stream>>>(qkh, qk_conv, nullptr, qkhid, 128, ROWS * 128);
  k_rotary<<<(unsigned)ROWS, 128, 0, stream>>>(qkhid, gamma, beta, qqf, qqbf, lqbf, qkf, lkT);
  k_pack_BT<<<blocks(ROWS * 128), 256, 0, stream>>>(qkf, qk_p, 128, 256, ROWS * 128);

  // ---- quad attention: attn = ReLU(qq @ qk^T / 256)^2 ; [qv|qu] = attn @ hid ----
  k_gemm<2, 0><<<dim3(256 / 128, 256 / 64, 128), 256, 0, stream>>>(
      qqbf, 256 * 128, qk_p, 128 * 256, nullptr, attn, 256 * 256,
      256, 128, 256, 1.0f / 256.0f);
  k_cvt_bf16<<<blocks(128LL * 256 * 256), 256, 0, stream>>>(attn, attnbf, 128LL * 256 * 256);
  k_gemm<0, 0><<<dim3(2048 / 128, 256 / 64, 128), 256, 0, stream>>>(
      attnbf, 256 * 256, hid_p, 256 * 2048, nullptr, attuv, 256 * 2048,
      256, 256, 2048, 1.0f);

  // ---- cross-chunk attention over T=64 (accumulates into attuv) ----
  k_cross<<<dim3(64, 256, 2), 256, 0, stream>>>(qqf, qkf, hid, attuv);

  // ---- linear attention: linkv = lin_k^T @ hid / 256 ; attuv += lin_q @ linkv ----
  k_gemm<0, 0><<<dim3(2048 / 128, 128 / 64, 128), 256, 0, stream>>>(
      lkT, 128 * 256, hid_p, 256 * 2048, nullptr, linkv, 128 * 2048,
      128, 256, 2048, 1.0f / 256.0f);
  k_pack_B<<<blocks(128LL * 128 * 2048), 256, 0, stream>>>(linkv, linkvp, 128, 2048,
                                                           128LL * 128 * 2048);
  k_gemm<0, 1><<<dim3(2048 / 128, 256 / 64, 128), 256, 0, stream>>>(
      lqbf, 256 * 128, linkvp, 128 * 2048, nullptr, attuv, 256 * 2048,
      256, 128, 2048, 1.0f);

  // ---- gating + out LN -> bf16; out FFN; final residual + dwconv ----
  k_gate_ln<<<(unsigned)ROWS, 256, 0, stream>>>(attuv, hid, o_g, o_bln, Aout);
  k_gemm<1, 0><<<dim3(512 / 128, (unsigned)(ROWS / 64), 1), 256, 0, stream>>>(
      Aout, 0, outWp, 0, out_b, oh, 0, (int)ROWS, 1024, 512, 1.0f);
  k_dwconv<<<blocks(ROWS * 512), 256, 0, stream>>>(oh, o_conv, x, out, 512, ROWS * 512);
}
